// MixtralDecoderLayerRoutable_50869592655194
// MI455X (gfx1250) — matrix operand order repaired and checked by reading, not compiled, not run on hardware
//
#include <hip/hip_runtime.h>
#include <hip/hip_bf16.h>
#include <math.h>

// ---------------------------------------------------------------------------
// Types for CDNA5 WMMA (wave32, v_wmma_f32_16x16x32_bf16)
// ---------------------------------------------------------------------------
typedef __bf16 bf16;
typedef __attribute__((ext_vector_type(16))) __bf16 v16bf;
typedef __attribute__((ext_vector_type(8)))  __bf16 v8bf;
typedef __attribute__((ext_vector_type(8)))  float  v8f;
typedef unsigned int u32x4 __attribute__((ext_vector_type(4)));
typedef int i32x4 __attribute__((ext_vector_type(4)));
typedef int i32x8 __attribute__((ext_vector_type(8)));
// async-to-LDS builtin wants int4* in global (AS1) / LDS (AS3) address spaces
typedef __attribute__((address_space(1))) i32x4* gptr_i32x4;
typedef __attribute__((address_space(3))) i32x4* lptr_i32x4;

#define SEQ   2048
#define HDIM  1024
#define NHEAD 16
#define KVH   8
#define HD    64
#define NEXP  8
#define IDIM  3584
#define EPSV  1e-6f

// ---- CDNA5 async / TDM feature probes (device pass only) -------------------
#if !defined(__HIP_DEVICE_COMPILE__)
  #define HAS_ASYNC 0
  #define HAS_TDM   0
#else
  #if __has_builtin(__builtin_amdgcn_global_load_async_to_lds_b128)
    #define HAS_ASYNC 1
  #else
    #define HAS_ASYNC 0
    #warning "CDNA5 probe: __builtin_amdgcn_global_load_async_to_lds_b128 NOT available -> ds_store fallback"
  #endif
  #if __has_builtin(__builtin_amdgcn_tensor_load_to_lds) && __has_builtin(__builtin_amdgcn_s_wait_tensorcnt)
    #define HAS_TDM 1
  #else
    #define HAS_TDM 0
    #warning "CDNA5 probe: tensor_load_to_lds / s_wait_tensorcnt NOT available"
  #endif
#endif

#if HAS_ASYNC
  #if __has_builtin(__builtin_amdgcn_s_wait_asynccnt)
    #define WAIT_ASYNC(n) __builtin_amdgcn_s_wait_asynccnt(n)
  #else
    #define WAIT_ASYNC(n) asm volatile("s_wait_asynccnt %0" :: "i"(n) : "memory")
  #endif
#else
  #define WAIT_ASYNC(n) ((void)0)
#endif

#if HAS_TDM
  #define WAIT_TENSOR(n) __builtin_amdgcn_s_wait_tensorcnt(n)
  #define ASYNC_PER_TILE 4   /* A tile only; B via TDM */
#else
  #define WAIT_TENSOR(n) ((void)0)
  #define ASYNC_PER_TILE 6   /* A (4) + B (2) async loads per thread */
#endif

__device__ __forceinline__ unsigned lds_offset_of(const void* p) {
#if defined(__HIP_DEVICE_COMPILE__)
  return (unsigned)(unsigned long long)(__attribute__((address_space(3))) const void*)p;
#else
  return 0;
#endif
}

__device__ __forceinline__ v8f wmma_bf16(v16bf a, v16bf b, v8f c) {
  return __builtin_amdgcn_wmma_f32_16x16x32_bf16(
      /*neg_a=*/false, a, /*neg_b=*/false, b,
      /*c_mod=*/(short)0, c, /*reuse_a=*/false, /*reuse_b=*/false);
}

// A-fragment (16x32): element j <-> k = half*16 + j
__device__ __forceinline__ v16bf load_a_frag(const bf16* __restrict__ row,
                                             int k0, int half) {
  v8bf lo = *(const v8bf*)(row + k0 + half * 16);
  v8bf hi = *(const v8bf*)(row + k0 + half * 16 + 8);
  v16bf a;
#pragma unroll
  for (int j = 0; j < 8; ++j) { a[j] = lo[j]; a[8 + j] = hi[j]; }
  return a;
}

#if HAS_TDM
// 2-D TDM descriptor: load a 64(rows) x 64(cols) bf16 tile of B[N,K] into LDS.
__device__ __forceinline__ void tdm_load_tile_b(const bf16* Bg, int gn0, int k0,
                                                int K, int N, unsigned lds_byte_off) {
  unsigned long long ga =
      (unsigned long long)(size_t)(const void*)(Bg + (size_t)gn0 * K + k0);
  u32x4 g0;
  g0[0] = 1u;                                           // count=1 (valid user D#)
  g0[1] = lds_byte_off;                                 // lds_addr
  g0[2] = (unsigned)(ga & 0xffffffffu);                 // global_addr[31:0]
  g0[3] = (unsigned)((ga >> 32) & 0x01ffffffu) | (2u << 30);  // addr[56:32] | type=2
  i32x8 g1;
  g1[0] = (int)(1u << 16);                              // data_size = 2 bytes
  g1[1] = (int)(((unsigned)K & 0xffffu) << 16);         // tensor_dim0[15:0]
  g1[2] = (int)((((unsigned)K >> 16) & 0xffffu) |
                (((unsigned)N & 0xffffu) << 16));       // dim0[31:16] | dim1[15:0]
  g1[3] = (int)((((unsigned)N >> 16) & 0xffffu) |
                (64u << 16));                           // dim1[31:16] | tile_dim0=64
  g1[4] = 64;                                           // tile_dim1=64, tile_dim2=0
  g1[5] = K;                                            // tensor_dim0_stride[31:0]
  g1[6] = 0;                                            // stride hi / dim1_stride lo
  g1[7] = 0;
  i32x4 z4 = {0, 0, 0, 0};
#if defined(__clang_major__) && __clang_major__ >= 23
  i32x8 z8 = {0, 0, 0, 0, 0, 0, 0, 0};
  __builtin_amdgcn_tensor_load_to_lds(g0, g1, z4, z4, z8, 0);
#else
  __builtin_amdgcn_tensor_load_to_lds(g0, g1, z4, z4, 0);
#endif
}
#endif

// ---------------------------------------------------------------------------
// Block-cooperative GEMM: C[M,N] f32 = A[M,K] bf16 * B[N,K]^T bf16.
// Block tile 128x64, BK=64, double-buffered LDS; A staged via async-to-LDS,
// B staged via TDM (fallbacks: async, then plain ds stores).
// ---------------------------------------------------------------------------
#define BM 128
#define BN 64
#define BK 64

__global__ void __launch_bounds__(256)
gemm_bf16(const bf16* __restrict__ A, const bf16* __restrict__ B,
          float* __restrict__ C, int M, int N, int K) {
  __shared__ __align__(32) bf16 As[2][BM * BK];
  __shared__ __align__(32) bf16 Bs[2][BN * BK];
  const int t = threadIdx.x;
  const int lane = t & 31;
  const int wslot = t >> 5;
  const int ntn = N / BN;
  const int bm = blockIdx.x / ntn, bn = blockIdx.x % ntn;
  const int gm0 = bm * BM, gn0 = bn * BN;
  const int m = lane & 15, half = lane >> 4;
  const int chunk = (t & 7) * 8;   // 8-elem (16B) column chunk
  const int srow = t >> 3;         // staging row 0..31

  auto stage = [&](int k0, int buf) {
#if HAS_ASYNC
#pragma unroll
    for (int r = 0; r < 4; ++r) {
      const bf16* src = A + (size_t)(gm0 + srow + r * 32) * K + k0 + chunk;
      bf16* dst = &As[buf][(srow + r * 32) * BK + chunk];
      __builtin_amdgcn_global_load_async_to_lds_b128((gptr_i32x4)src,
                                                     (lptr_i32x4)dst, 0, 0);
    }
#if !HAS_TDM
#pragma unroll
    for (int r = 0; r < 2; ++r) {
      const bf16* src = B + (size_t)(gn0 + srow + r * 32) * K + k0 + chunk;
      bf16* dst = &Bs[buf][(srow + r * 32) * BK + chunk];
      __builtin_amdgcn_global_load_async_to_lds_b128((gptr_i32x4)src,
                                                     (lptr_i32x4)dst, 0, 0);
    }
#endif
#else   // synchronous fallback staging (ds_store_b128)
#pragma unroll
    for (int r = 0; r < 4; ++r)
      *(v8bf*)&As[buf][(srow + r * 32) * BK + chunk] =
          *(const v8bf*)(A + (size_t)(gm0 + srow + r * 32) * K + k0 + chunk);
#if !HAS_TDM
#pragma unroll
    for (int r = 0; r < 2; ++r)
      *(v8bf*)&Bs[buf][(srow + r * 32) * BK + chunk] =
          *(const v8bf*)(B + (size_t)(gn0 + srow + r * 32) * K + k0 + chunk);
#endif
#endif
#if HAS_TDM
    if (wslot == 0)
      tdm_load_tile_b(B, gn0, k0, K, N, lds_offset_of(&Bs[buf][0]));
#endif
  };

  v8f acc0 = {}, acc1 = {}, acc2 = {}, acc3 = {};
  stage(0, 0);
  const int nk = K / BK;
  for (int kt = 0; kt < nk; ++kt) {
    const int cur = kt & 1;
    if (kt + 1 < nk) {
      stage((kt + 1) * BK, cur ^ 1);
      WAIT_ASYNC(ASYNC_PER_TILE);          // leave next tile's loads in flight
#if HAS_TDM
      if (wslot == 0) WAIT_TENSOR(1);
#endif
    } else {
      WAIT_ASYNC(0);
#if HAS_TDM
      if (wslot == 0) WAIT_TENSOR(0);
#endif
    }
    __syncthreads();
    const bf16* Abuf = &As[cur][(wslot * 16 + m) * BK];
    const bf16* Bbuf = &Bs[cur][0];
#pragma unroll
    for (int ks = 0; ks < BK; ks += 32) {
      v16bf a = load_a_frag(Abuf, ks, half);
      v16bf b0 = *(const v16bf*)(Bbuf + (0 * 16 + m) * BK + ks + half * 16);
      v16bf b1 = *(const v16bf*)(Bbuf + (1 * 16 + m) * BK + ks + half * 16);
      v16bf b2 = *(const v16bf*)(Bbuf + (2 * 16 + m) * BK + ks + half * 16);
      v16bf b3 = *(const v16bf*)(Bbuf + (3 * 16 + m) * BK + ks + half * 16);
      acc0 = wmma_bf16(a, b0, acc0);
      acc1 = wmma_bf16(a, b1, acc1);
      acc2 = wmma_bf16(a, b2, acc2);
      acc3 = wmma_bf16(a, b3, acc3);
    }
    __syncthreads();
  }
  const int r0 = gm0 + wslot * 16;
#pragma unroll
  for (int r = 0; r < 8; ++r) {
    float* Cr = C + (size_t)(r0 + r + half * 8) * N + gn0 + m;
    Cr[0]  = acc0[r];
    Cr[16] = acc1[r];
    Cr[32] = acc2[r];
    Cr[48] = acc3[r];
  }
}

// ---------------------------------------------------------------------------
// RMSNorm (f32 in) -> bf16 out. One 256-thread block per row.
// ---------------------------------------------------------------------------
__global__ void rmsnorm_bf16(const float* __restrict__ x, const float* __restrict__ w,
                             bf16* __restrict__ out, int H) {
  __shared__ float red[8];
  __shared__ float stot;
  const int row = blockIdx.x;
  const float* xr = x + (size_t)row * H;
  float ss = 0.f;
  for (int j = threadIdx.x; j < H; j += blockDim.x) { float v = xr[j]; ss += v * v; }
#pragma unroll
  for (int sh = 16; sh >= 1; sh >>= 1) ss += __shfl_xor(ss, sh, 32);
  if ((threadIdx.x & 31) == 0) red[threadIdx.x >> 5] = ss;
  __syncthreads();
  if (threadIdx.x == 0) {
    float tsum = 0.f;
#pragma unroll
    for (int i = 0; i < 8; ++i) tsum += red[i];
    stot = tsum;
  }
  __syncthreads();
  const float inv = rsqrtf(stot / (float)H + EPSV);
  for (int j = threadIdx.x; j < H; j += blockDim.x)
    out[(size_t)row * H + j] = (bf16)(xr[j] * inv * w[j]);
}

// ---------------------------------------------------------------------------
// RoPE + repack: qb [NH][S][HD] (q pre-scaled 1/sqrt(HD)), kb [KVH][S][HD],
// vt [KVH][HD][S] (transposed for the P*V B-fragment).
// ---------------------------------------------------------------------------
__global__ void rope_pack(const float* __restrict__ qtmp, const float* __restrict__ ktmp,
                          const float* __restrict__ vtmp, bf16* __restrict__ qb,
                          bf16* __restrict__ kb, bf16* __restrict__ vt) {
  const int s = blockIdx.x;
  const float* q = qtmp + (size_t)s * (NHEAD * HD);
  const float* k = ktmp + (size_t)s * (KVH * HD);
  const float* v = vtmp + (size_t)s * (KVH * HD);

  for (int idx = threadIdx.x; idx < NHEAD * 32; idx += blockDim.x) {
    int h = idx >> 5, i = idx & 31;
    float f = (float)s * __powf(1.0e6f, -(float)(2 * i) / 64.0f);
    float c = __cosf(f), sn = __sinf(f);
    float x0 = q[h * HD + i], x1 = q[h * HD + 32 + i];
    bf16* dst = qb + ((size_t)h * SEQ + s) * HD;
    dst[i]      = (bf16)((x0 * c - x1 * sn) * 0.125f);
    dst[32 + i] = (bf16)((x1 * c + x0 * sn) * 0.125f);
  }
  for (int idx = threadIdx.x; idx < KVH * 32; idx += blockDim.x) {
    int h = idx >> 5, i = idx & 31;
    float f = (float)s * __powf(1.0e6f, -(float)(2 * i) / 64.0f);
    float c = __cosf(f), sn = __sinf(f);
    float x0 = k[h * HD + i], x1 = k[h * HD + 32 + i];
    bf16* dst = kb + ((size_t)h * SEQ + s) * HD;
    dst[i]      = (bf16)(x0 * c - x1 * sn);
    dst[32 + i] = (bf16)(x1 * c + x0 * sn);
  }
  for (int idx = threadIdx.x; idx < KVH * HD; idx += blockDim.x) {
    int h = idx >> 6, d = idx & 63;
    vt[((size_t)h * HD + d) * SEQ + s] = (bf16)v[h * HD + d];
  }
}

// ---------------------------------------------------------------------------
// Flash attention (causal, GQA rep=2). One wave per (head, 16-query block).
// ---------------------------------------------------------------------------
__global__ void flash_attn(const bf16* __restrict__ Qb, const bf16* __restrict__ Kb,
                           const bf16* __restrict__ Vt, bf16* __restrict__ Ctx) {
  __shared__ __align__(32) bf16 pshare[4][16 * 32];
  const int lane = threadIdx.x & 31;
  const int wslot = threadIdx.x >> 5;
  const int gwave = blockIdx.x * 4 + wslot;
  const int nQB = SEQ >> 4;
  const int h = gwave / nQB;
  if (h >= NHEAD) return;
  const int qb = gwave % nQB;
  const int kvh = h >> 1;
  const int q0 = qb << 4;
  const int m = lane & 15, half = lane >> 4;

  const bf16* Qrow = Qb + ((size_t)h * SEQ + q0 + m) * HD;
  v16bf qf0 = load_a_frag(Qrow, 0, half);
  v16bf qf1 = load_a_frag(Qrow, 32, half);

  v8f acc0 = {}, acc1 = {}, acc2 = {}, acc3 = {};
  float mrun[8], lrun[8];
#pragma unroll
  for (int r = 0; r < 8; ++r) { mrun[r] = -3.0e38f; lrun[r] = 0.f; }

  const bf16* Kbase = Kb + (size_t)kvh * SEQ * HD;
  const bf16* Vbase = Vt + (size_t)kvh * HD * SEQ;
  const int nTB = (q0 + 47) >> 5;

  for (int tb = 0; tb < nTB; ++tb) {
    const int t0 = tb << 5;
    v8f s0 = {}, s1 = {};
    {
      v16bf b00 = *(const v16bf*)(Kbase + (size_t)(t0 + m) * HD + 0 + half * 16);
      v16bf b01 = *(const v16bf*)(Kbase + (size_t)(t0 + 16 + m) * HD + 0 + half * 16);
      s0 = wmma_bf16(qf0, b00, s0);
      s1 = wmma_bf16(qf0, b01, s1);
      v16bf b10 = *(const v16bf*)(Kbase + (size_t)(t0 + m) * HD + 32 + half * 16);
      v16bf b11 = *(const v16bf*)(Kbase + (size_t)(t0 + 16 + m) * HD + 32 + half * 16);
      s0 = wmma_bf16(qf1, b10, s0);
      s1 = wmma_bf16(qf1, b11, s1);
    }
    if (t0 + 31 > q0) {
#pragma unroll
      for (int r = 0; r < 8; ++r) {
        int qrow = q0 + r + half * 8;
        if (t0 + m > qrow)      s0[r] = -3.0e38f;
        if (t0 + 16 + m > qrow) s1[r] = -3.0e38f;
      }
    }
#pragma unroll
    for (int r = 0; r < 8; ++r) {
      float rv = fmaxf(s0[r], s1[r]);
#pragma unroll
      for (int sh = 8; sh >= 1; sh >>= 1) rv = fmaxf(rv, __shfl_xor(rv, sh, 32));
      float mnew = fmaxf(mrun[r], rv);
      float scale = __expf(mrun[r] - mnew);
      float p0 = __expf(s0[r] - mnew);
      float p1 = __expf(s1[r] - mnew);
      float rs = p0 + p1;
#pragma unroll
      for (int sh = 8; sh >= 1; sh >>= 1) rs += __shfl_xor(rs, sh, 32);
      lrun[r] = lrun[r] * scale + rs;
      mrun[r] = mnew;
      acc0[r] *= scale; acc1[r] *= scale; acc2[r] *= scale; acc3[r] *= scale;
      pshare[wslot][(r + half * 8) * 32 + m]      = (bf16)p0;
      pshare[wslot][(r + half * 8) * 32 + 16 + m] = (bf16)p1;
    }
    v16bf pf = load_a_frag(&pshare[wslot][m * 32], 0, half);
#pragma unroll
    for (int i = 0; i < 4; ++i) {
      v16bf vf = *(const v16bf*)(Vbase + (size_t)(i * 16 + m) * SEQ + t0 + half * 16);
      v8f* accp = (i == 0) ? &acc0 : (i == 1) ? &acc1 : (i == 2) ? &acc2 : &acc3;
      *accp = wmma_bf16(pf, vf, *accp);
    }
  }
#pragma unroll
  for (int r = 0; r < 8; ++r) {
    float inv = 1.0f / lrun[r];
    bf16* out = Ctx + (size_t)(q0 + r + half * 8) * (NHEAD * HD) + h * HD;
    out[0 + m]  = (bf16)(acc0[r] * inv);
    out[16 + m] = (bf16)(acc1[r] * inv);
    out[32 + m] = (bf16)(acc2[r] * inv);
    out[48 + m] = (bf16)(acc3[r] * inv);
  }
}

// ---------------------------------------------------------------------------
// Small elementwise / routing kernels
// ---------------------------------------------------------------------------
__global__ void f32_to_bf16(const float* __restrict__ in, bf16* __restrict__ out, size_t n) {
  size_t i = (size_t)blockIdx.x * blockDim.x + threadIdx.x;
  if (i < n) out[i] = (bf16)in[i];
}

__global__ void add_f32(const float* __restrict__ a, const float* __restrict__ b,
                        float* __restrict__ c, size_t n) {
  size_t i = (size_t)blockIdx.x * blockDim.x + threadIdx.x;
  if (i < n) c[i] = a[i] + b[i];
}

__global__ void silu_mul_bf16(const float* __restrict__ h1, const float* __restrict__ h3,
                              bf16* __restrict__ hb, size_t n) {
  size_t i = (size_t)blockIdx.x * blockDim.x + threadIdx.x;
  if (i < n) {
    float a = h1[i];
    float s = a / (1.0f + __expf(-a));
    hb[i] = (bf16)(s * h3[i]);
  }
}

__global__ void accum_expert(float* __restrict__ moe, const float* __restrict__ tmp,
                             const float* __restrict__ wm, int e, int first, size_t n) {
  size_t i = (size_t)blockIdx.x * blockDim.x + threadIdx.x;
  if (i < n) {
    size_t tok = i / HDIM;
    float w = wm[tok * NEXP + e];
    float base = first ? 0.0f : moe[i];
    moe[i] = base + w * tmp[i];
  }
}

// Top-2 routing (argmax over all 28 pairs == two largest softmax entries).
__global__ void route_top2(const bf16* __restrict__ xn, const float* __restrict__ gate,
                           float* __restrict__ wm) {
  __shared__ float logits[NEXP];
  const int n = blockIdx.x;
  const int wv = threadIdx.x >> 5, lane = threadIdx.x & 31;
  const bf16* xr = xn + (size_t)n * HDIM;
  const float* g = gate + (size_t)wv * HDIM;
  float acc = 0.f;
  for (int j = lane; j < HDIM; j += 32) acc += (float)xr[j] * g[j];
#pragma unroll
  for (int sh = 16; sh >= 1; sh >>= 1) acc += __shfl_xor(acc, sh, 32);
  if (lane == 0) logits[wv] = acc;
  __syncthreads();
  if (threadIdx.x == 0) {
    float mx = -3.0e38f;
    for (int e = 0; e < NEXP; ++e) mx = fmaxf(mx, logits[e]);
    float rw[NEXP], sum = 0.f;
    for (int e = 0; e < NEXP; ++e) { rw[e] = __expf(logits[e] - mx); sum += rw[e]; }
    for (int e = 0; e < NEXP; ++e) rw[e] /= sum;
    int a = 0;
    for (int e = 1; e < NEXP; ++e) if (rw[e] > rw[a]) a = e;
    int b = (a == 0) ? 1 : 0;
    for (int e = 0; e < NEXP; ++e) if (e != a && rw[e] > rw[b]) b = e;
    float ws = rw[a] + rw[b];
    for (int e = 0; e < NEXP; ++e) wm[(size_t)n * NEXP + e] = 0.f;
    wm[(size_t)n * NEXP + a] = rw[a] / ws;
    wm[(size_t)n * NEXP + b] = rw[b] / ws;
  }
}

// ---------------------------------------------------------------------------
// Host-side orchestration
// ---------------------------------------------------------------------------
static inline void launch_gemm(const bf16* A, const bf16* B, float* C,
                               int M, int N, int K, hipStream_t s) {
  int blocks = (M / BM) * (N / BN);
  gemm_bf16<<<blocks, 256, 0, s>>>(A, B, C, M, N, K);
}
static inline void launch_conv(const float* in, bf16* out, size_t n, hipStream_t s) {
  f32_to_bf16<<<(unsigned)((n + 255) / 256), 256, 0, s>>>(in, out, n);
}

extern "C" void kernel_launch(void* const* d_in, const int* in_sizes, int n_in,
                              void* d_out, int out_size, void* d_ws, size_t ws_size,
                              hipStream_t stream) {
  const float* hid  = (const float*)d_in[0];
  const float* ln1  = (const float*)d_in[1];
  const float* ln2  = (const float*)d_in[2];
  const float* wq   = (const float*)d_in[3];
  const float* wk   = (const float*)d_in[4];
  const float* wv   = (const float*)d_in[5];
  const float* wo   = (const float*)d_in[6];
  const float* gate = (const float*)d_in[7];
  const float* w1   = (const float*)d_in[8];
  const float* w2   = (const float*)d_in[9];
  const float* w3   = (const float*)d_in[10];
  float* out = (float*)d_out;

  char* p = (char*)d_ws;
  auto take = [&](size_t bytes) -> char* {
    char* r = p;
    p += (bytes + 255) & ~(size_t)255;
    return r;
  };

  const size_t NTOK = SEQ;                      // B==1
  bf16* wq_b = (bf16*)take((size_t)HDIM * HDIM * 2);
  bf16* wk_b = (bf16*)take((size_t)KVH * HD * HDIM * 2);
  bf16* wv_b = (bf16*)take((size_t)KVH * HD * HDIM * 2);
  bf16* wo_b = (bf16*)take((size_t)HDIM * HDIM * 2);
  bf16* w1_b = (bf16*)take((size_t)NEXP * IDIM * HDIM * 2);
  bf16* w2_b = (bf16*)take((size_t)NEXP * HDIM * IDIM * 2);
  bf16* w3_b = (bf16*)take((size_t)NEXP * IDIM * HDIM * 2);
  bf16* xn1  = (bf16*)take((size_t)NTOK * HDIM * 2);
  float* qtmp = (float*)take((size_t)NTOK * NHEAD * HD * 4);
  float* ktmp = (float*)take((size_t)NTOK * KVH * HD * 4);
  float* vtmp = (float*)take((size_t)NTOK * KVH * HD * 4);
  bf16* qb   = (bf16*)take((size_t)NHEAD * SEQ * HD * 2);
  bf16* kb   = (bf16*)take((size_t)KVH * SEQ * HD * 2);
  bf16* vt   = (bf16*)take((size_t)KVH * HD * SEQ * 2);
  bf16* ctx  = (bf16*)take((size_t)NTOK * NHEAD * HD * 2);
  float* attn = (float*)take((size_t)NTOK * HDIM * 4);
  float* hid2 = (float*)take((size_t)NTOK * HDIM * 4);
  bf16* xn2  = (bf16*)take((size_t)NTOK * HDIM * 2);
  float* wmr  = (float*)take((size_t)NTOK * NEXP * 4);
  float* h1   = (float*)take((size_t)NTOK * IDIM * 4);
  float* h3   = (float*)take((size_t)NTOK * IDIM * 4);
  bf16* hb   = (bf16*)take((size_t)NTOK * IDIM * 2);
  float* etmp = (float*)take((size_t)NTOK * HDIM * 4);
  float* moe  = (float*)take((size_t)NTOK * HDIM * 4);

  // Weight conversion to bf16 (one-time per launch; << WMMA time)
  launch_conv(wq, wq_b, (size_t)HDIM * HDIM, stream);
  launch_conv(wk, wk_b, (size_t)KVH * HD * HDIM, stream);
  launch_conv(wv, wv_b, (size_t)KVH * HD * HDIM, stream);
  launch_conv(wo, wo_b, (size_t)HDIM * HDIM, stream);
  launch_conv(w1, w1_b, (size_t)NEXP * IDIM * HDIM, stream);
  launch_conv(w2, w2_b, (size_t)NEXP * HDIM * IDIM, stream);
  launch_conv(w3, w3_b, (size_t)NEXP * IDIM * HDIM, stream);

  // --- Attention ---
  rmsnorm_bf16<<<SEQ, 256, 0, stream>>>(hid, ln1, xn1, HDIM);
  launch_gemm(xn1, wq_b, qtmp, SEQ, NHEAD * HD, HDIM, stream);
  launch_gemm(xn1, wk_b, ktmp, SEQ, KVH * HD, HDIM, stream);
  launch_gemm(xn1, wv_b, vtmp, SEQ, KVH * HD, HDIM, stream);
  rope_pack<<<SEQ, 256, 0, stream>>>(qtmp, ktmp, vtmp, qb, kb, vt);
  {
    int waves = NHEAD * (SEQ / 16);
    flash_attn<<<(waves + 3) / 4, 128, 0, stream>>>(qb, kb, vt, ctx);
  }
  launch_gemm(ctx, wo_b, attn, SEQ, HDIM, HDIM, stream);
  add_f32<<<(unsigned)((NTOK * HDIM + 255) / 256), 256, 0, stream>>>(hid, attn, hid2,
                                                                     NTOK * HDIM);

  // --- MoE ---
  rmsnorm_bf16<<<SEQ, 256, 0, stream>>>(hid2, ln2, xn2, HDIM);
  route_top2<<<SEQ, 256, 0, stream>>>(xn2, gate, wmr);
  for (int e = 0; e < NEXP; ++e) {
    const bf16* w1e = w1_b + (size_t)e * IDIM * HDIM;
    const bf16* w3e = w3_b + (size_t)e * IDIM * HDIM;
    const bf16* w2e = w2_b + (size_t)e * HDIM * IDIM;
    launch_gemm(xn2, w1e, h1, SEQ, IDIM, HDIM, stream);
    launch_gemm(xn2, w3e, h3, SEQ, IDIM, HDIM, stream);
    silu_mul_bf16<<<(unsigned)((NTOK * IDIM + 255) / 256), 256, 0, stream>>>(
        h1, h3, hb, NTOK * IDIM);
    launch_gemm(hb, w2e, etmp, SEQ, HDIM, IDIM, stream);
    accum_expert<<<(unsigned)((NTOK * HDIM + 255) / 256), 256, 0, stream>>>(
        moe, etmp, wmr, e, e == 0 ? 1 : 0, NTOK * HDIM);
  }
  add_f32<<<(unsigned)((NTOK * HDIM + 255) / 256), 256, 0, stream>>>(hid2, moe, out,
                                                                     NTOK * HDIM);
}
